// GPT2ParallelTransformer_84739704750803
// MI455X (gfx1250) — matrix operand order repaired and checked by reading, not compiled, not run on hardware
//
#include <hip/hip_runtime.h>
#include <hip/hip_bf16.h>

// ---------------------------------------------------------------------------
// GPT-2 transformer (L=2, B=1, S=2048, H=1024, N=16 heads, HN=64) for gfx1250.
// All GEMMs run on V_WMMA_F32_16X16X32_BF16 (bf16 inputs, fp32 accumulate).
// Attention is flash-style (no SxS score materialization).
// LDS staging uses GLOBAL_LOAD_ASYNC_TO_LDS_B128 (ASYNCcnt-tracked DMA path).
// ---------------------------------------------------------------------------

typedef __attribute__((ext_vector_type(16))) __bf16        v16bf;
typedef __attribute__((ext_vector_type(8)))  float         v8f;
typedef __attribute__((ext_vector_type(4)))  unsigned int  v4u;
typedef __attribute__((ext_vector_type(4)))  int           v4i;

#define SS   2048
#define HH   1024
#define H3   3072
#define H4   4096
#define NHD  16

#define GLOBAL_AS __attribute__((address_space(1)))
#define LDS_AS    __attribute__((address_space(3)))

// --- async copy global->LDS (CDNA5), guarded so compilation never breaks ----
#if __has_builtin(__builtin_amdgcn_global_load_async_to_lds_b128)
#define ASYNC_CP_B128(gsrc, ldst)                                              \
    __builtin_amdgcn_global_load_async_to_lds_b128(                            \
        (GLOBAL_AS v4i*)(gsrc), (LDS_AS v4i*)(ldst), 0, 0)
#else
#define ASYNC_CP_B128(gsrc, ldst) (*(v4u*)(ldst) = *(const v4u*)(gsrc))
#endif

#if __has_builtin(__builtin_amdgcn_s_wait_asynccnt)
#define ASYNC_WAIT() __builtin_amdgcn_s_wait_asynccnt(0)
#else
#define ASYNC_WAIT() asm volatile("s_wait_asynccnt 0" ::: "memory")
#endif

static __device__ __forceinline__ unsigned short f2bf(float f) {
    unsigned int u = __float_as_uint(f);
    u = u + 0x7fffu + ((u >> 16) & 1u);     // round-to-nearest-even
    return (unsigned short)(u >> 16);
}

struct FragBF {
    union { v16bf bf; v4u u[2]; };
};

// A-matrix 16x32 bf16 fragment (ISA 7.12.2):
// lane<16 : row M = lane,    K chunks {0..7, 16..23}
// lane>=16: row M = lane-16, K chunks {8..15, 24..31}
static __device__ __forceinline__ FragBF loadA(const unsigned short* base, int stride, int lane) {
    const int r  = lane & 15;
    const int ko = (lane < 16) ? 0 : 8;
    const unsigned short* p = base + r * stride + ko;
    FragBF f;
    f.u[0] = *(const v4u*)(p);
    f.u[1] = *(const v4u*)(p + 16);
    return f;
}

// B-matrix 32x16 bf16 fragment: lane holds column n = lane&15;
// lane<16 holds K {0..7, 16..23}, lane>=16 holds K {8..15, 24..31}.
static __device__ __forceinline__ FragBF loadB(const unsigned short* base, int stride, int lane) {
    const int c  = lane & 15;
    const int ko = (lane < 16) ? 0 : 8;
    const unsigned short* p = base + c * stride + ko;
    FragBF f;
    f.u[0] = *(const v4u*)(p);
    f.u[1] = *(const v4u*)(p + 16);
    return f;
}

static __device__ __forceinline__ v8f wmma_bf16(const FragBF& a, const FragBF& b, v8f c) {
    return __builtin_amdgcn_wmma_f32_16x16x32_bf16(false, a.bf, false, b.bf,
                                                   (short)0, c, false, false);
}

static __device__ __forceinline__ float gelu_f(float v) {
    float t = 0.7978845608028654f * v * (1.0f + 0.044715f * v * v);
    return 0.5f * v * (1.0f + tanhf(t));
}

// ---------------------------------------------------------------------------
// Cast + transpose weights: W[k][n] fp32 -> Wt[n][k] bf16
// ---------------------------------------------------------------------------
__global__ __launch_bounds__(256) void cast_tr_kernel(const float* __restrict__ W,
                                                      unsigned short* __restrict__ Wt,
                                                      int K, int Nn) {
    int id = blockIdx.x * 256 + threadIdx.x;
    if (id >= K * Nn) return;
    int n = id / K;
    int k = id - n * K;
    Wt[id] = f2bf(W[k * Nn + n]);
}

// ---------------------------------------------------------------------------
// LayerNorm over H=1024; writes bf16 (outb) or fp32 (outf).
// ---------------------------------------------------------------------------
__global__ __launch_bounds__(256) void ln_kernel(const float* __restrict__ x,
                                                 const float* __restrict__ g,
                                                 const float* __restrict__ b,
                                                 unsigned short* __restrict__ outb,
                                                 float* __restrict__ outf) {
    __shared__ float red[256];
    __shared__ float red2[256];
    const int row = blockIdx.x;
    const int tid = threadIdx.x;
    const float* xr = x + row * HH;

    float v[4], s = 0.f, s2 = 0.f;
    for (int i = 0; i < 4; ++i) {
        v[i] = xr[tid + i * 256];
        s  += v[i];
        s2 += v[i] * v[i];
    }
    red[tid] = s; red2[tid] = s2;
    __syncthreads();
    for (int st = 128; st > 0; st >>= 1) {
        if (tid < st) { red[tid] += red[tid + st]; red2[tid] += red2[tid + st]; }
        __syncthreads();
    }
    const float mu   = red[0]  * (1.0f / HH);
    const float var  = red2[0] * (1.0f / HH) - mu * mu;
    const float rstd = rsqrtf(var + 1e-5f);
    for (int i = 0; i < 4; ++i) {
        int j = tid + i * 256;
        float y = (v[i] - mu) * rstd * g[j] + b[j];
        if (outb) outb[row * HH + j] = f2bf(y);
        else      outf[row * HH + j] = y;
    }
}

// ---------------------------------------------------------------------------
// GEMM: out[M,N] = A[M,K](bf16) * W[K,N] + bias, W stored transposed Wt[N,K].
// mode 0: store bf16 ; mode 1: gelu then store bf16 ; mode 2: fp32 residual +=
// Block = 128 threads (4 wave32), block tile 128x64, wave tile 32x64, BK=32.
// Per wave per K-step: 2 A frags + 4 B frags -> 8 WMMAs (B frags reused 2x).
// ---------------------------------------------------------------------------
__global__ __launch_bounds__(128) void gemm_wmma_bf16_kernel(
    const unsigned short* __restrict__ A,
    const unsigned short* __restrict__ Wt,
    const float* __restrict__ bias,
    unsigned short* __restrict__ outb,
    float* __restrict__ outf,
    int M, int N, int K, int mode)
{
    __shared__ __align__(16) unsigned short As[128][32];
    __shared__ __align__(16) unsigned short Bs[64][32];

    const int tid  = threadIdx.x;
    const int lane = tid & 31;
    const int w    = tid >> 5;
    const int n0   = blockIdx.x * 64;
    const int m0   = blockIdx.y * 128;

    v8f acc[2][4];
    for (int i = 0; i < 2; ++i)
        for (int j = 0; j < 4; ++j)
            for (int r = 0; r < 8; ++r) acc[i][j][r] = 0.f;

    for (int k0 = 0; k0 < K; k0 += 32) {
        // stage A tile 128x32 (512 x b128 chunks) and B tile 64x32 (256 chunks)
        for (int c = tid; c < 512; c += 128) {
            const int row = c >> 2;
            const int q   = (c & 3) * 8;
            ASYNC_CP_B128(&A[(m0 + row) * K + k0 + q], &As[row][q]);
        }
        for (int c = tid; c < 256; c += 128) {
            const int row = c >> 2;
            const int q   = (c & 3) * 8;
            ASYNC_CP_B128(&Wt[(n0 + row) * K + k0 + q], &Bs[row][q]);
        }
        ASYNC_WAIT();
        __syncthreads();

        FragBF a0 = loadA(&As[w * 32 +  0][0], 32, lane);
        FragBF a1 = loadA(&As[w * 32 + 16][0], 32, lane);
        for (int ng = 0; ng < 4; ++ng) {
            FragBF b = loadB(&Bs[ng * 16][0], 32, lane);
            acc[0][ng] = wmma_bf16(a0, b, acc[0][ng]);
            acc[1][ng] = wmma_bf16(a1, b, acc[1][ng]);
        }
        __syncthreads();
    }

    // C layout: lane -> n = lane&15 ; VGPR r -> m = r + (lane<16 ? 0 : 8)
    const int ncol = n0 + (lane & 15);
    for (int i = 0; i < 2; ++i) {
        const int mrow = m0 + w * 32 + i * 16 + ((lane < 16) ? 0 : 8);
        for (int ng = 0; ng < 4; ++ng) {
            const int n = ncol + ng * 16;
            const float bv = bias ? bias[n] : 0.f;
            for (int r = 0; r < 8; ++r) {
                const int m = mrow + r;
                float v = acc[i][ng][r] + bv;
                if (mode == 1) v = gelu_f(v);
                if (mode == 2) outf[m * N + n] += v;    // residual h += (A*W + b)
                else           outb[m * N + n]  = f2bf(v);
            }
        }
    }
}

// ---------------------------------------------------------------------------
// Flash attention: one block = (head, 64-query strip), 128 threads / 4 waves.
// q/k/v read from fused qkv (bf16, ld=3072); ctx written bf16 [S,H].
// ---------------------------------------------------------------------------
__global__ __launch_bounds__(128) void attn_flash_kernel(
    const unsigned short* __restrict__ qkv,
    unsigned short* __restrict__ ctx)
{
    __shared__ __align__(16) unsigned short Ks[32][64];   // keys-major, d contiguous
    __shared__ __align__(16) unsigned short Vt[64][32];   // d-major, keys contiguous
    __shared__ __align__(16) unsigned short Pl[4][16][32];

    const int tid  = threadIdx.x;
    const int lane = tid & 31;
    const int w    = tid >> 5;
    const int hh   = blockIdx.y;
    const int qb   = blockIdx.x * 64;
    const int ldq  = H3;

    // Q strip (16 rows per wave) as two A fragments: d 0..31 and d 32..63
    const unsigned short* qbase = qkv + (qb + w * 16) * ldq + hh * 64;
    FragBF aq0 = loadA(qbase +  0, ldq, lane);
    FragBF aq1 = loadA(qbase + 32, ldq, lane);

    v8f o[4];
    for (int i = 0; i < 4; ++i)
        for (int j = 0; j < 8; ++j) o[i][j] = 0.f;
    float mrun[8], lrun[8];
    for (int r = 0; r < 8; ++r) { mrun[r] = -1e30f; lrun[r] = 0.f; }

    const int nkt = (qb + 64) >> 5;          // key tiles of 32, causal-truncated
    for (int kt = 0; kt < nkt; ++kt) {
        const int kk0 = kt * 32;
        // K tile: async straight to LDS; V tile: via VGPRs (transposed store)
        for (int c = tid; c < 256; c += 128) {
            const int row = c >> 3;
            const int q   = (c & 7) * 8;
            ASYNC_CP_B128(&qkv[(kk0 + row) * ldq + HH + hh * 64 + q], &Ks[row][q]);
            union { v4u u; unsigned short s[8]; } t;
            t.u = *(const v4u*)&qkv[(kk0 + row) * ldq + 2 * HH + hh * 64 + q];
            for (int i = 0; i < 8; ++i) Vt[q + i][row] = t.s[i];
        }
        ASYNC_WAIT();
        __syncthreads();

        // scores: S[16q x 32k] = Q[16x64] * K^T  (two C frags, contraction d=64)
        v8f s0, s1;
        for (int j = 0; j < 8; ++j) { s0[j] = 0.f; s1[j] = 0.f; }
        s0 = wmma_bf16(aq0, loadB(&Ks[ 0][ 0], 64, lane), s0);
        s0 = wmma_bf16(aq1, loadB(&Ks[ 0][32], 64, lane), s0);
        s1 = wmma_bf16(aq0, loadB(&Ks[16][ 0], 64, lane), s1);
        s1 = wmma_bf16(aq1, loadB(&Ks[16][32], 64, lane), s1);

        const int key0  = kk0 + (lane & 15);
        const int mbase = qb + w * 16 + ((lane < 16) ? 0 : 8);
        const int m_in  = (lane < 16) ? 0 : 8;

        for (int r = 0; r < 8; ++r) {
            const int mg = mbase + r;
            float v0 = s0[r] * 0.125f; if (key0      > mg) v0 = -10000.f;
            float v1 = s1[r] * 0.125f; if (key0 + 16 > mg) v1 = -10000.f;

            // row max across the 16 lanes of this half-wave (xor masks stay in half)
            float mx = fmaxf(v0, v1);
            for (int msk = 1; msk < 16; msk <<= 1)
                mx = fmaxf(mx, __shfl_xor(mx, msk, 32));
            const float mnew = fmaxf(mrun[r], mx);
            const float corr = __expf(mrun[r] - mnew);
            mrun[r] = mnew;

            const float p0 = __expf(v0 - mnew);
            const float p1 = __expf(v1 - mnew);
            float sum = p0 + p1;
            for (int msk = 1; msk < 16; msk <<= 1)
                sum += __shfl_xor(sum, msk, 32);
            lrun[r] = lrun[r] * corr + sum;
            for (int ng = 0; ng < 4; ++ng) o[ng][r] *= corr;

            // C layout -> A layout via per-wave LDS patch
            Pl[w][m_in + r][(lane & 15)]      = f2bf(p0);
            Pl[w][m_in + r][16 + (lane & 15)] = f2bf(p1);
        }

        // ctx += P[16x32] * V[32x64]
        FragBF ap = loadA(&Pl[w][0][0], 32, lane);
        for (int ng = 0; ng < 4; ++ng) {
            FragBF bv = loadB(&Vt[ng * 16][0], 32, lane);
            o[ng] = wmma_bf16(ap, bv, o[ng]);
        }
        __syncthreads();
    }

    const int mrow = qb + w * 16 + ((lane < 16) ? 0 : 8);
    for (int ng = 0; ng < 4; ++ng) {
        const int col = hh * 64 + ng * 16 + (lane & 15);
        for (int r = 0; r < 8; ++r)
            ctx[(mrow + r) * HH + col] = f2bf(o[ng][r] / lrun[r]);
    }
}

// ---------------------------------------------------------------------------
// Host-side orchestration
// ---------------------------------------------------------------------------
extern "C" void kernel_launch(void* const* d_in, const int* in_sizes, int n_in,
                              void* d_out, int out_size, void* d_ws, size_t ws_size,
                              hipStream_t stream) {
    (void)in_sizes; (void)n_in; (void)out_size; (void)ws_size;

    const float* x_in    = (const float*)d_in[0];
    // d_in[1] = ltor_mask (causality implemented analytically)
    const float* ln1_g   = (const float*)d_in[2];
    const float* ln1_b   = (const float*)d_in[3];
    const float* qkv_w   = (const float*)d_in[4];
    const float* qkv_b   = (const float*)d_in[5];
    const float* dense_w = (const float*)d_in[6];
    const float* dense_b = (const float*)d_in[7];
    const float* ln2_g   = (const float*)d_in[8];
    const float* ln2_b   = (const float*)d_in[9];
    const float* fc_w    = (const float*)d_in[10];
    const float* fc_b    = (const float*)d_in[11];
    const float* proj_w  = (const float*)d_in[12];
    const float* proj_b  = (const float*)d_in[13];
    const float* lnf_g   = (const float*)d_in[14];
    const float* lnf_b   = (const float*)d_in[15];

    char* ws = (char*)d_ws;
    size_t off = 0;
    float*          h     = (float*)(ws + off);          off += (size_t)SS * HH * 4;
    unsigned short* ybf   = (unsigned short*)(ws + off); off += (size_t)SS * HH * 2;
    unsigned short* qkvbf = (unsigned short*)(ws + off); off += (size_t)SS * H3 * 2;
    unsigned short* ctxbf = (unsigned short*)(ws + off); off += (size_t)SS * HH * 2;
    unsigned short* fcbf  = (unsigned short*)(ws + off); off += (size_t)SS * H4 * 2;
    unsigned short* wQt   = (unsigned short*)(ws + off); off += (size_t)H3 * HH * 2;
    unsigned short* wDt   = (unsigned short*)(ws + off); off += (size_t)HH * HH * 2;
    unsigned short* wFt   = (unsigned short*)(ws + off); off += (size_t)H4 * HH * 2;
    unsigned short* wPt   = (unsigned short*)(ws + off); off += (size_t)HH * H4 * 2;

    (void)hipMemcpyAsync(h, x_in, (size_t)SS * HH * 4, hipMemcpyDeviceToDevice, stream);

    for (int l = 0; l < 2; ++l) {
        // weights -> bf16, transposed (N-major)
        cast_tr_kernel<<<(HH * H3 + 255) / 256, 256, 0, stream>>>(
            qkv_w + (size_t)l * HH * H3, wQt, HH, H3);
        cast_tr_kernel<<<(HH * HH + 255) / 256, 256, 0, stream>>>(
            dense_w + (size_t)l * HH * HH, wDt, HH, HH);
        cast_tr_kernel<<<(HH * H4 + 255) / 256, 256, 0, stream>>>(
            fc_w + (size_t)l * HH * H4, wFt, HH, H4);
        cast_tr_kernel<<<(H4 * HH + 255) / 256, 256, 0, stream>>>(
            proj_w + (size_t)l * H4 * HH, wPt, H4, HH);

        // y = LN1(h)
        ln_kernel<<<SS, 256, 0, stream>>>(h, ln1_g + l * HH, ln1_b + l * HH, ybf, nullptr);

        // qkv = y @ qkv_w + qkv_b  (bf16 out)
        gemm_wmma_bf16_kernel<<<dim3(H3 / 64, SS / 128), 128, 0, stream>>>(
            ybf, wQt, qkv_b + l * H3, qkvbf, nullptr, SS, H3, HH, 0);

        // ctx = softmax(qk^T/8, causal) @ v
        attn_flash_kernel<<<dim3(SS / 64, NHD), 128, 0, stream>>>(qkvbf, ctxbf);

        // h = h + ctx @ dense_w + dense_b  (fp32 residual)
        gemm_wmma_bf16_kernel<<<dim3(HH / 64, SS / 128), 128, 0, stream>>>(
            ctxbf, wDt, dense_b + l * HH, nullptr, h, SS, HH, HH, 2);

        // y2 = LN2(h)
        ln_kernel<<<SS, 256, 0, stream>>>(h, ln2_g + l * HH, ln2_b + l * HH, ybf, nullptr);

        // fcact = gelu(y2 @ fc_w + fc_b)  (bf16 out)
        gemm_wmma_bf16_kernel<<<dim3(H4 / 64, SS / 128), 128, 0, stream>>>(
            ybf, wFt, fc_b + l * H4, fcbf, nullptr, SS, H4, HH, 1);

        // h = h + fcact @ proj_w + proj_b
        gemm_wmma_bf16_kernel<<<dim3(HH / 64, SS / 128), 128, 0, stream>>>(
            fcbf, wPt, proj_b + l * HH, nullptr, h, SS, HH, H4, 2);
    }

    // out = LN_f(h), fp32
    ln_kernel<<<SS, 256, 0, stream>>>(h, lnf_g, lnf_b, nullptr, (float*)d_out);
}
